// IneqOptNet_42365557408299
// MI455X (gfx1250) — matrix operand order repaired and checked by reading, not compiled, not run on hardware
//
#include <hip/hip_runtime.h>
#include <stdint.h>

typedef __attribute__((ext_vector_type(4)))  float  v4f;
typedef __attribute__((ext_vector_type(8)))  float  v8f;
typedef __attribute__((ext_vector_type(8)))  __bf16 v8bf;
typedef __attribute__((ext_vector_type(16))) __bf16 v16bf;

constexpr int B_   = 2048;
constexpr int NF_  = 1024;
constexpr int NH_  = 2048;
constexpr int NC_  = 128;
constexpr int M_   = 20;
constexpr int N2G  = M_ * NC_;   // 2560 columns handled by WMMA
constexpr int NY_  = N2G + M_;   // 2580 rows of W2
constexpr int QP_ITERS  = 500;
constexpr int POW_ITERS = 64;

// ---------------------------------------------------------------------------
// Split f32 -> (hi, lo) bf16 pair: v ~= (f32)hi + (f32)lo  (bf16x3 scheme)
// ---------------------------------------------------------------------------
__global__ __launch_bounds__(256) void split_f32_bf16(
    const float* __restrict__ src, __bf16* __restrict__ hi,
    __bf16* __restrict__ lo, int n4)
{
  const int i = blockIdx.x * blockDim.x + threadIdx.x;
  if (i >= n4) return;
  v4f v = ((const v4f*)src)[i];
#pragma unroll
  for (int c = 0; c < 4; ++c) {
    float f = v[c];
    __bf16 h = (__bf16)f;
    __bf16 l = (__bf16)(f - (float)h);
    hi[i * 4 + c] = h;
    lo[i * 4 + c] = l;
  }
}

// Load 16 bf16 elements as two contiguous 8-element (16B) chunks
__device__ __forceinline__ v16bf ld16(const __bf16* p, int stride) {
  v8bf x0 = *(const v8bf*)p;
  v8bf x1 = *(const v8bf*)(p + stride);
  return __builtin_shufflevector(x0, x1, 0, 1, 2, 3, 4, 5, 6, 7,
                                 8, 9, 10, 11, 12, 13, 14, 15);
}

// ---------------------------------------------------------------------------
// Block-tiled bf16x3 GEMM: C[M x N] = A[M x K] * W[N x K]^T + bias.
// Block = 8 waves = 128(M) x 64(N); all waves share one 64-col B strip.
// B chunk (64 cols x 64 K, hi+lo) is staged into LDS in WMMA-fragment image
// layout via GLOBAL_LOAD_ASYNC_TO_LDS_B128 (ASYNCcnt), double-buffered:
//   frag(t, k32, h) = 32 lanes x 32B;  writer lane puts its 32B at lane*32;
//   reader: two ds_load_b128 at +0/+16.
// A rows are private per wave -> direct global b128 loads.
// MODE 1: relu + re-split output to (Chi, Clo).   MODE 0: f32 output to Cf.
// ---------------------------------------------------------------------------
template <int MODE>
__global__ __launch_bounds__(256) void gemm_bf16x3(
    const __bf16* __restrict__ Ahi, const __bf16* __restrict__ Alo,
    const __bf16* __restrict__ Whi, const __bf16* __restrict__ Wlo,
    const float* __restrict__ bias,
    float* __restrict__ Cf, __bf16* __restrict__ Chi, __bf16* __restrict__ Clo,
    int Ndim, int K)
{
  // 2 buffers x 16 frags x 512 bf16 (1KB) = 32 KB
  __shared__ __align__(16) __bf16 Bs[2 * 16 * 512];

  const int wave = threadIdx.x >> 5;
  const int lane = threadIdx.x & 31;
  const int nbn  = Ndim >> 6;
  const int bm   = blockIdx.x / nbn;
  const int bn   = blockIdx.x % nbn;
  const int tm   = (bm << 7) + (wave << 4);
  const int tn0  = bn << 6;
  const int half = lane >> 4;
  const int l    = lane & 15;

  // --- A fragment addressing (per-wave private rows) ---
  const size_t aoff = (size_t)(tm + l) * K + (half << 4);
  const __bf16* arh = Ahi + aoff;
  const __bf16* arl = Alo + aoff;

  // --- B staging: wave w owns frag pair (t = w>>1, k32 = w&1, h = hi/lo) ---
  const int st = wave >> 1;
  const int sk = wave & 1;
  const uint32_t vbase = (uint32_t)(((size_t)(tn0 + (st << 4) + l) * K
                                     + (sk << 5) + (half << 4)) * 2);
  const int sfrag = ((st << 1) | sk) << 1;                    // hi frag index
  const uint32_t lds_w0 = (uint32_t)(uintptr_t)(&Bs[0] + sfrag * 512 + lane * 16);
  const unsigned long long sWhi = (unsigned long long)(uintptr_t)Whi;
  const unsigned long long sWlo = (unsigned long long)(uintptr_t)Wlo;

  const int nck = K >> 6;            // chunks of KC=64
  v8f acc[4] = {};

  // prologue: stage chunk 0 into buffer 0
  {
    uint32_t va = vbase;
    uint32_t lh = lds_w0, ll = lds_w0 + 1024;
    asm volatile(
        "global_load_async_to_lds_b128 %0, %2, %3 offset:0\n\t"
        "global_load_async_to_lds_b128 %0, %2, %3 offset:16\n\t"
        "global_load_async_to_lds_b128 %1, %2, %4 offset:0\n\t"
        "global_load_async_to_lds_b128 %1, %2, %4 offset:16"
        :: "v"(lh), "v"(ll), "v"(va), "s"(sWhi), "s"(sWlo)
        : "memory");
  }

  for (int ci = 0; ci < nck; ++ci) {
    const int buf = ci & 1;
    const bool has_next = (ci + 1) < nck;
    if (has_next) {
      uint32_t va = vbase + (uint32_t)((ci + 1) << 7);       // +64 K * 2B
      uint32_t lh = lds_w0 + (uint32_t)((buf ^ 1) * 16384);
      uint32_t ll = lh + 1024;
      asm volatile(
          "global_load_async_to_lds_b128 %0, %2, %3 offset:0\n\t"
          "global_load_async_to_lds_b128 %0, %2, %3 offset:16\n\t"
          "global_load_async_to_lds_b128 %1, %2, %4 offset:0\n\t"
          "global_load_async_to_lds_b128 %1, %2, %4 offset:16"
          :: "v"(lh), "v"(ll), "v"(va), "s"(sWhi), "s"(sWlo)
          : "memory");
      asm volatile("s_wait_asynccnt 0x4" ::: "memory");      // chunk ci landed
    } else {
      asm volatile("s_wait_asynccnt 0x0" ::: "memory");
    }
    __syncthreads();   // all waves' fragments visible

    const __bf16* bufp = &Bs[buf * 8192];
#pragma unroll
    for (int k32 = 0; k32 < 2; ++k32) {
      const int k0 = (ci << 6) + (k32 << 5);
      v16bf ah = ld16(arh + k0, 8);
      v16bf al = ld16(arl + k0, 8);
#pragma unroll
      for (int t = 0; t < 4; ++t) {
        const __bf16* fb = bufp + ((((t << 1) | k32) << 1) * 512) + lane * 16;
        v16bf bh = ld16(fb, 8);
        v16bf bl = ld16(fb + 512, 8);
        acc[t] = __builtin_amdgcn_wmma_f32_16x16x32_bf16(
            false, ah, false, bh, (short)0, acc[t], false, false);
        acc[t] = __builtin_amdgcn_wmma_f32_16x16x32_bf16(
            false, ah, false, bl, (short)0, acc[t], false, false);
        acc[t] = __builtin_amdgcn_wmma_f32_16x16x32_bf16(
            false, al, false, bh, (short)0, acc[t], false, false);
      }
    }
    __syncthreads();   // done reading buf before it is overwritten
  }

#pragma unroll
  for (int t = 0; t < 4; ++t) {
    const int col = tn0 + (t << 4) + l;
    const float bv = bias[col];
#pragma unroll
    for (int v = 0; v < 8; ++v) {
      const int m = v + (half << 3);
      const size_t idx = (size_t)(tm + m) * Ndim + col;
      float val = acc[t][v] + bv;
      if constexpr (MODE == 1) {
        val = fmaxf(val, 0.f);
        __bf16 h = (__bf16)val;
        Chi[idx] = h;
        Clo[idx] = (__bf16)(val - (float)h);
      } else {
        Cf[idx] = val;
      }
    }
  }
}

// ---------------------------------------------------------------------------
// Tail columns of y: h[b][j] = dot(h1[b,:], W2[2560+j,:]) + b2[2560+j]
// ---------------------------------------------------------------------------
__global__ __launch_bounds__(256) void tail_cols(
    const __bf16* __restrict__ h1hi, const __bf16* __restrict__ h1lo,
    const float* __restrict__ W2, const float* __restrict__ b2,
    float* __restrict__ hb)
{
  const int idx = blockIdx.x * blockDim.x + threadIdx.x;  // b*20 + j
  const int b = idx / M_;
  const int j = idx % M_;
  const v8bf* ah = (const v8bf*)(h1hi + (size_t)b * NH_);
  const v8bf* al = (const v8bf*)(h1lo + (size_t)b * NH_);
  const v4f*  w  = (const v4f*)(W2 + (size_t)(N2G + j) * NH_);
  float s = 0.f;
  for (int k = 0; k < NH_ / 8; ++k) {
    v8bf hv = ah[k], lv = al[k];
    v4f w0 = w[2 * k], w1 = w[2 * k + 1];
#pragma unroll
    for (int c = 0; c < 4; ++c) {
      s += ((float)hv[c] + (float)lv[c]) * w0[c];
      s += ((float)hv[4 + c] + (float)lv[4 + c]) * w1[c];
    }
  }
  hb[idx] = s + b2[N2G + j];
}

// ---------------------------------------------------------------------------
// Per-batch QP (FISTA, 500 iters) + log_softmax. One wave32 per batch row.
// ---------------------------------------------------------------------------
__global__ __launch_bounds__(128) void qp_softmax(
    const float* __restrict__ Gbuf, const float* __restrict__ hb,
    float* __restrict__ out)
{
  __shared__ float Gs[4][M_ * NC_];
  __shared__ float lamS[4][M_];
  const int wave = threadIdx.x >> 5;
  const int lane = threadIdx.x & 31;
  const int b = blockIdx.x * 4 + wave;

  float* Gw = Gs[wave];
  const float* gsrc = Gbuf + (size_t)b * N2G;
  for (int i = lane; i < M_ * NC_; i += 32) Gw[i] = gsrc[i];
  __syncthreads();

  float gg[M_];
#pragma unroll
  for (int j = 0; j < M_; ++j) gg[j] = 0.f;
  float cvec = 0.f;
  const bool act = lane < M_;
  if (act) {
    const float* gi = Gw + lane * NC_;
    for (int j = 0; j < M_; ++j) {
      const float* gj = Gw + j * NC_;
      float s = 0.f;
      for (int n = 0; n < NC_; ++n) s += gi[n] * gj[n];
      gg[j] = s;
    }
    float csum = 0.f;
    for (int n = 0; n < NC_; ++n) csum += gi[n];
    cvec = csum + hb[b * M_ + lane];
  }

  // Power iteration: lambda_max(GG)  (GG PSD => spectral norm)
  float v = act ? 1.f : 0.f;
  float lambda = 0.f;
  for (int it = 0; it < POW_ITERS; ++it) {
    float w = 0.f;
#pragma unroll
    for (int j = 0; j < M_; ++j) w += gg[j] * __shfl(v, j, 32);
    float s2 = w * w;
#pragma unroll
    for (int off = 16; off; off >>= 1) s2 += __shfl_xor(s2, off, 32);
    float nrm = sqrtf(s2);
    lambda = nrm;
    v = (nrm > 0.f) ? (w / nrm) : 0.f;
  }
  const float invL = 1.f / (lambda * 1.0001f + 1e-6f);

  // FISTA
  float lam = 0.f, yv = 0.f, t = 1.f;
  for (int it = 0; it < QP_ITERS; ++it) {
    float grad = cvec;
#pragma unroll
    for (int j = 0; j < M_; ++j) grad += gg[j] * __shfl(yv, j, 32);
    float lam_new = fmaxf(yv - grad * invL, 0.f);
    float t_new = 0.5f * (1.f + sqrtf(1.f + 4.f * t * t));
    yv = lam_new + ((t - 1.f) / t_new) * (lam_new - lam);
    lam = lam_new;
    t = t_new;
  }
  if (act) lamS[wave][lane] = lam;
  __syncthreads();

  float z[4];
#pragma unroll
  for (int p = 0; p < 4; ++p) {
    const int n = lane + (p << 5);
    float s = 0.f;
#pragma unroll
    for (int i = 0; i < M_; ++i) s += Gw[i * NC_ + n] * lamS[wave][i];
    z[p] = -(1.f + s);
  }
  float mx = fmaxf(fmaxf(z[0], z[1]), fmaxf(z[2], z[3]));
#pragma unroll
  for (int off = 16; off; off >>= 1) mx = fmaxf(mx, __shfl_xor(mx, off, 32));
  float se = 0.f;
#pragma unroll
  for (int p = 0; p < 4; ++p) se += __expf(z[p] - mx);
#pragma unroll
  for (int off = 16; off; off >>= 1) se += __shfl_xor(se, off, 32);
  const float lse = __logf(se);
  float* ob = out + (size_t)b * NC_;
#pragma unroll
  for (int p = 0; p < 4; ++p) ob[lane + (p << 5)] = z[p] - mx - lse;
}

// ---------------------------------------------------------------------------
extern "C" void kernel_launch(void* const* d_in, const int* in_sizes, int n_in,
                              void* d_out, int out_size, void* d_ws, size_t ws_size,
                              hipStream_t stream) {
  const float* x  = (const float*)d_in[0];
  const float* W1 = (const float*)d_in[1];
  const float* b1 = (const float*)d_in[2];
  const float* W2 = (const float*)d_in[3];
  const float* b2 = (const float*)d_in[4];
  float* out = (float*)d_out;

  char* ws = (char*)d_ws;
  const size_t sz_x  = (size_t)B_ * NF_ * 2;
  const size_t sz_w1 = (size_t)NH_ * NF_ * 2;
  const size_t sz_w2 = (size_t)NY_ * NH_ * 2;
  const size_t sz_h1 = (size_t)B_ * NH_ * 2;
  size_t off = 0;
  __bf16* x_hi  = (__bf16*)(ws + off); off += sz_x;
  __bf16* x_lo  = (__bf16*)(ws + off); off += sz_x;
  __bf16* w1_hi = (__bf16*)(ws + off); off += sz_w1;
  __bf16* w1_lo = (__bf16*)(ws + off); off += sz_w1;
  __bf16* w2_hi = (__bf16*)(ws + off); off += sz_w2;
  __bf16* w2_lo = (__bf16*)(ws + off); off += sz_w2;
  __bf16* h1_hi = (__bf16*)(ws + off); off += sz_h1;
  __bf16* h1_lo = (__bf16*)(ws + off); off += sz_h1;
  float*  Gbuf  = (float*)(ws + off);  off += (size_t)B_ * N2G * 4;
  float*  hbuf  = (float*)(ws + off);

  split_f32_bf16<<<(B_ * NF_ / 4) / 256, 256, 0, stream>>>(x, x_hi, x_lo, B_ * NF_ / 4);
  split_f32_bf16<<<(NH_ * NF_ / 4) / 256, 256, 0, stream>>>(W1, w1_hi, w1_lo, NH_ * NF_ / 4);
  split_f32_bf16<<<(NY_ * NH_ / 4) / 256, 256, 0, stream>>>(W2, w2_hi, w2_lo, NY_ * NH_ / 4);

  // h1 = relu(x @ W1^T + b1), emitted as bf16 hi/lo split
  gemm_bf16x3<1><<<(B_ / 128) * (NH_ / 64), 256, 0, stream>>>(
      x_hi, x_lo, w1_hi, w1_lo, b1, nullptr, h1_hi, h1_lo, NH_, NF_);

  // G = (h1 @ W2^T + b2)[:, :2560] in f32
  gemm_bf16x3<0><<<(B_ / 128) * (N2G / 64), 256, 0, stream>>>(
      h1_hi, h1_lo, w2_hi, w2_lo, b2, Gbuf, nullptr, nullptr, N2G, NH_);

  // h = (h1 @ W2^T + b2)[:, 2560:]
  tail_cols<<<(B_ * M_) / 256, 256, 0, stream>>>(h1_hi, h1_lo, W2, b2, hbuf);

  // QP solve + log_softmax
  qp_softmax<<<B_ / 4, 128, 0, stream>>>(Gbuf, hbuf, out);
}
